// HybridDnCNNWithPreprocessing_11484742549768
// MI455X (gfx1250) — matrix operand order repaired and checked by reading, not compiled, not run on hardware
//
#include <hip/hip_runtime.h>
#include <hip/hip_bf16.h>

// ---------------------------------------------------------------------------
// HybridDnCNNWithPreprocessing for MI455X (gfx1250, wave32, WMMA).
// Channel contractions run on v_wmma_f32_16x16x32_f16 (fp16 in, fp32 acc).
// ---------------------------------------------------------------------------

typedef _Float16 half_t;
typedef __attribute__((ext_vector_type(16))) _Float16 v16h;
typedef __attribute__((ext_vector_type(8)))  float    v8f;

#define B_   2
#define H_   256
#define W_   256
#define HW_  (H_ * W_)          // 65536
#define P_   (B_ * HW_)         // 131072 pixels
#define F_   64
#define C3F_ 192
#define BN_EPS 1e-5f

union Frag16 { v16h v; half_t h[16]; float4 f4[2]; };
union Acc8   { v8f  v; float  f[8]; };

// ========================== preprocess (bilateral) =========================
__device__ __forceinline__ float blended_at(const float* __restrict__ x,
                                            int b, int yy, int xx) {
  // reflect padding (no edge repeat), radius 2
  if (yy < 0) yy = -yy;           if (yy >= H_) yy = 2 * H_ - 2 - yy;
  if (xx < 0) xx = -xx;           if (xx >= W_) xx = 2 * W_ - 2 - xx;
  float g = (x[(long)b * HW_ + yy * W_ + xx] + 1.0f) * 0.5f;
  g = fminf(fmaxf(g, 1e-6f), 1.0f);
  return 0.6f * g + 0.4f * powf(g, 1.5f);   // (1-ALPHA)g + ALPHA g^GAMMA
}

__global__ void preprocess_kernel(const float* __restrict__ x,
                                  float* __restrict__ xp) {
  long p = (long)blockIdx.x * blockDim.x + threadIdx.x;
  if (p >= P_) return;
  int b = (int)(p >> 16), rem = (int)(p & (HW_ - 1));
  int y = rem >> 8, xx = rem & 255;
  float c = blended_at(x, b, y, xx);
  float num = 0.f, den = 0.f;
  const float inv_sc2 = 1.0f / (0.05f * 0.05f);
  const float inv_ss2 = 1.0f / (50.0f * 50.0f);
#pragma unroll
  for (int di = -2; di <= 2; ++di)
#pragma unroll
    for (int dj = -2; dj <= 2; ++dj) {
      if (di * di + dj * dj > 4) continue;
      float sh = blended_at(x, b, y + di, xx + dj);
      float sw = __expf(-0.5f * (float)(di * di + dj * dj) * inv_ss2);
      float d  = sh - c;
      float wg = sw * __expf(-0.5f * d * d * inv_sc2);
      num += wg * sh; den += wg;
    }
  xp[p] = num / den;
}

// ================= first multi-scale conv: 1ch -> 192ch + relu ==============
__global__ void conv_in_kernel(const float* __restrict__ xp,
                               const float* __restrict__ w3,
                               const float* __restrict__ w5,
                               const float* __restrict__ w7,
                               half_t* __restrict__ h1) {   // [P][192]
  long idx = (long)blockIdx.x * blockDim.x + threadIdx.x;
  if (idx >= (long)P_ * C3F_) return;
  int co = (int)(idx % C3F_);
  long p = idx / C3F_;
  int b = (int)(p >> 16), rem = (int)(p & (HW_ - 1));
  int y = rem >> 8, xx = rem & 255;
  const float* w; int ks, c;
  if (co < 64)       { w = w3; ks = 3; c = co; }
  else if (co < 128) { w = w5; ks = 5; c = co - 64; }
  else               { w = w7; ks = 7; c = co - 128; }
  int R = ks >> 1;
  float acc = 0.f;
  for (int ky = 0; ky < ks; ++ky) {
    int yy = y + ky - R; if (yy < 0 || yy >= H_) continue;
    for (int kx = 0; kx < ks; ++kx) {
      int xw = xx + kx - R; if (xw < 0 || xw >= W_) continue;
      acc += xp[(long)b * HW_ + yy * W_ + xw] * w[c * ks * ks + ky * ks + kx];
    }
  }
  h1[idx] = (half_t)fmaxf(acc, 0.f);
}

// ======================== weight repack (f32 -> f16) =======================
__global__ void pack_f16_kernel(const float* __restrict__ src,
                                half_t* __restrict__ dst, int n) {
  int i = blockIdx.x * blockDim.x + threadIdx.x;
  if (i < n) dst[i] = (half_t)src[i];
}

// OIHW [64][64][k][k] -> [tap][co][ci] so B-fragment loads are contiguous
__global__ void pack_conv_kernel(const float* __restrict__ src,
                                 half_t* __restrict__ dst, int ks) {
  int kk = ks * ks, n = F_ * F_ * kk;
  int i = blockIdx.x * blockDim.x + threadIdx.x;
  if (i >= n) return;
  int co = i / (F_ * kk), rem = i % (F_ * kk);
  int ci = rem / kk,      t   = rem % kk;
  dst[((long)t * F_ + co) * F_ + ci] = (half_t)src[i];
}

// ================== WMMA GEMM: out[P,64] = A[P,K] * W[64,K]^T ==============
// Block: 128 threads = 4 waves; wave w handles out-channels [16w,16w+16).
__global__ void gemm_pw_kernel(const half_t* __restrict__ A,
                               const half_t* __restrict__ Wt,
                               void* __restrict__ outp,
                               int K, int out_f16) {
  const int lane = threadIdx.x & 31;
  const int wv   = threadIdx.x >> 5;
  const int r    = lane & 15;
  const int hi   = lane >> 4;
  const long tileM  = (long)blockIdx.x * 16;
  const int  cobase = wv * 16;

  Acc8 acc;
#pragma unroll
  for (int i = 0; i < 8; ++i) acc.f[i] = 0.f;

  for (int kb = 0; kb < K; kb += 32) {
    Frag16 a, bfr;
    // A frag: lane holds row M=r, K = kb + hi*8 + [0..7] and kb+16+hi*8+[0..7]
    const half_t* ap = A + (tileM + r) * K + kb + hi * 8;
    a.f4[0] = *reinterpret_cast<const float4*>(ap);
    a.f4[1] = *reinterpret_cast<const float4*>(ap + 16);
    // B frag: lane holds col N=r, K = kb + hi*8 + [0..7] and kb+16+hi*8+[0..7]
    const half_t* bp = Wt + (long)(cobase + r) * K + kb + hi * 8;
    bfr.f4[0] = *reinterpret_cast<const float4*>(bp);
    bfr.f4[1] = *reinterpret_cast<const float4*>(bp + 16);
    acc.v = __builtin_amdgcn_wmma_f32_16x16x32_f16(
        false, a.v, false, bfr.v, (short)0, acc.v, false, false);
  }
#pragma unroll
  for (int vv = 0; vv < 8; ++vv) {
    long row = tileM + vv + hi * 8;
    if (out_f16) ((half_t*)outp)[row * F_ + cobase + r] = (half_t)acc.f[vv];
    else         ((float* )outp)[row * F_ + cobase + r] = acc.f[vv];
  }
}

// =========== implicit-GEMM conv 64->64 via WMMA, relu, write slice =========
// X: NHWC f16 [P][64], Wp: [tap][co][ci] f16, Out: [P][192] (slice_off..+64)
__global__ void conv_gemm_kernel(const half_t* __restrict__ X,
                                 const half_t* __restrict__ Wp,
                                 half_t* __restrict__ Out,
                                 int ksize, int slice_off) {
  const int lane = threadIdx.x & 31;
  const int wv   = threadIdx.x >> 5;
  const int r    = lane & 15;
  const int hi   = lane >> 4;
  const int cobase = wv * 16;
  const long p0 = (long)blockIdx.x * 16;          // 16 pixels, one image row
  const int b   = (int)(p0 >> 16);
  const int rem = (int)(p0 & (HW_ - 1));
  const int y   = rem >> 8;
  const int x0  = rem & 255;
  const int R   = ksize >> 1;

  Acc8 acc;
#pragma unroll
  for (int i = 0; i < 8; ++i) acc.f[i] = 0.f;

  for (int ky = 0; ky < ksize; ++ky) {
    int yy = y + ky - R;
    bool yok = (yy >= 0) && (yy < H_);
    for (int kx = 0; kx < ksize; ++kx) {
      int xx = x0 + r + kx - R;
      bool ok = yok && (xx >= 0) && (xx < W_);
      long pix = (long)b * HW_ + (long)yy * W_ + xx;
      const half_t* abase = X + pix * F_ + hi * 8;
      if (ok && yy + 1 < H_)            // warm next tap row (global_prefetch)
        __builtin_prefetch(X + (pix + W_) * F_, 0, 1);
      const int tap = ky * ksize + kx;
      const half_t* bbase = Wp + ((long)tap * F_ + cobase + r) * F_ + hi * 8;
#pragma unroll
      for (int ch = 0; ch < 2; ++ch) {  // Cin = 64 -> two K=32 chunks
        Frag16 a, bfr;
        if (ok) {
          a.f4[0] = *reinterpret_cast<const float4*>(abase + ch * 32);
          a.f4[1] = *reinterpret_cast<const float4*>(abase + ch * 32 + 16);
        } else {
          a.f4[0] = make_float4(0.f, 0.f, 0.f, 0.f);
          a.f4[1] = make_float4(0.f, 0.f, 0.f, 0.f);
        }
        bfr.f4[0] = *reinterpret_cast<const float4*>(bbase + ch * 32);
        bfr.f4[1] = *reinterpret_cast<const float4*>(bbase + ch * 32 + 16);
        acc.v = __builtin_amdgcn_wmma_f32_16x16x32_f16(
            false, a.v, false, bfr.v, (short)0, acc.v, false, false);
      }
    }
  }
#pragma unroll
  for (int vv = 0; vv < 8; ++vv) {
    long row = p0 + vv + hi * 8;
    float t = acc.f[vv];
    Out[row * C3F_ + slice_off + cobase + r] = (half_t)(t > 0.f ? t : 0.f);
  }
}

// ============================ BN stats / apply =============================
__global__ void bn_stats_kernel(const float* __restrict__ m32,
                                float* __restrict__ mean,
                                float* __restrict__ var) {
  int c = blockIdx.x;                       // 64 blocks
  float s = 0.f, s2 = 0.f;
  for (long p = threadIdx.x; p < P_; p += blockDim.x) {
    float v = m32[p * F_ + c]; s += v; s2 += v * v;
  }
  __shared__ float sh[256], sh2[256];
  sh[threadIdx.x] = s; sh2[threadIdx.x] = s2;
  __syncthreads();
  for (int o = blockDim.x >> 1; o > 0; o >>= 1) {
    if ((int)threadIdx.x < o) { sh[threadIdx.x] += sh[threadIdx.x + o];
                                sh2[threadIdx.x] += sh2[threadIdx.x + o]; }
    __syncthreads();
  }
  if (threadIdx.x == 0) {
    float m = sh[0] / (float)P_;
    mean[c] = m; var[c] = sh2[0] / (float)P_ - m * m;
  }
}

__global__ void bn_apply_kernel(const float* __restrict__ m32,
                                const float* __restrict__ mean,
                                const float* __restrict__ var,
                                const float* __restrict__ g,
                                const float* __restrict__ bb,
                                half_t* __restrict__ mbn) {
  long i = (long)blockIdx.x * blockDim.x + threadIdx.x;
  if (i >= (long)P_ * F_) return;
  int c = (int)(i & (F_ - 1));
  float v = g[c] * (m32[i] - mean[c]) * rsqrtf(var[c] + BN_EPS) + bb[c];
  mbn[i] = (half_t)fmaxf(v, 0.f);
}

// ============================ SE (squeeze-excite) ==========================
__global__ void se_mean_kernel(const half_t* __restrict__ mbn,
                               float* __restrict__ ybar) {
  int b = blockIdx.x >> 6, c = blockIdx.x & 63;   // 128 blocks
  float s = 0.f;
  for (long i = threadIdx.x; i < HW_; i += blockDim.x)
    s += (float)mbn[((long)b * HW_ + i) * F_ + c];
  __shared__ float sh[256];
  sh[threadIdx.x] = s; __syncthreads();
  for (int o = blockDim.x >> 1; o > 0; o >>= 1) {
    if ((int)threadIdx.x < o) sh[threadIdx.x] += sh[threadIdx.x + o];
    __syncthreads();
  }
  if (threadIdx.x == 0) ybar[blockIdx.x] = sh[0] / (float)HW_;
}

__global__ void se_mlp_kernel(const float* __restrict__ ybar,
                              const float* __restrict__ se1,   // [4][64]
                              const float* __restrict__ se2,   // [64][4]
                              float* __restrict__ yse) {
  __shared__ float t[B_][4];
  int tid = threadIdx.x;                 // 128 threads
  int b = tid >> 6, c = tid & 63;
  if (c < 4) {
    float s = 0.f;
    for (int k = 0; k < F_; ++k) s += ybar[b * F_ + k] * se1[c * F_ + k];
    t[b][c] = fmaxf(s, 0.f);
  }
  __syncthreads();
  float s = 0.f;
#pragma unroll
  for (int j = 0; j < 4; ++j) s += t[b][j] * se2[c * 4 + j];
  yse[b * F_ + c] = 1.f / (1.f + __expf(-s));
}

// ============= final: noise = conv(m*y, wf);  out = xp - noise =============
__global__ void final_kernel(const float* __restrict__ xp,
                             const half_t* __restrict__ mbn,
                             const float* __restrict__ yse,
                             const float* __restrict__ wf,    // [64][3][3]
                             float* __restrict__ out) {
  __shared__ float ys[B_ * F_];
  if (threadIdx.x < B_ * F_) ys[threadIdx.x] = yse[threadIdx.x];
  __syncthreads();
  long p = (long)blockIdx.x * blockDim.x + threadIdx.x;
  if (p >= P_) return;
  int b = (int)(p >> 16), rem = (int)(p & (HW_ - 1));
  int y = rem >> 8, xx = rem & 255;
  float acc = 0.f;
  for (int ky = 0; ky < 3; ++ky) {
    int yy = y + ky - 1; if (yy < 0 || yy >= H_) continue;
    for (int kx = 0; kx < 3; ++kx) {
      int xw = xx + kx - 1; if (xw < 0 || xw >= W_) continue;
      long q = ((long)b * HW_ + yy * W_ + xw) * F_;
      int tap = ky * 3 + kx;
#pragma unroll 16
      for (int c = 0; c < F_; ++c)
        acc += (float)mbn[q + c] * ys[b * F_ + c] * wf[c * 9 + tap];
    }
  }
  out[p] = xp[p] - acc;
}

// =============================== launcher ==================================
extern "C" void kernel_launch(void* const* d_in, const int* in_sizes, int n_in,
                              void* d_out, int out_size, void* d_ws, size_t ws_size,
                              hipStream_t stream) {
  const float* x    = (const float*)d_in[0];
  const float* w_i3 = (const float*)d_in[1];
  const float* w_i5 = (const float*)d_in[2];
  const float* w_i7 = (const float*)d_in[3];
  const float* w_p1 = (const float*)d_in[4];
  const float* wb3  = (const float*)d_in[5];
  const float* wb5  = (const float*)d_in[6];
  const float* wb7  = (const float*)d_in[7];
  const float* wbp  = (const float*)d_in[8];
  const float* bn_g = (const float*)d_in[9];
  const float* bn_b = (const float*)d_in[10];
  const float* se1  = (const float*)d_in[11];
  const float* se2  = (const float*)d_in[12];
  const float* wf   = (const float*)d_in[13];
  float* out = (float*)d_out;

  // ---- workspace carve-up (all intermediates fit in MI455X's 192MB L2) ----
  char* ws = (char*)d_ws;
  size_t off = 0;
  auto carve = [&](size_t bytes) -> char* {
    char* p = ws + off; off = (off + bytes + 255) & ~(size_t)255; return p;
  };
  float*  xp    = (float*) carve((size_t)P_ * 4);
  half_t* buf192= (half_t*)carve((size_t)P_ * C3F_ * 2);  // h1, then mcat
  half_t* h2    = (half_t*)carve((size_t)P_ * F_ * 2);
  float*  m32   = (float*) carve((size_t)P_ * F_ * 4);
  half_t* mbn   = (half_t*)carve((size_t)P_ * F_ * 2);
  half_t* wp1h  = (half_t*)carve((size_t)F_ * C3F_ * 2);
  half_t* wbph  = (half_t*)carve((size_t)F_ * C3F_ * 2);
  half_t* wb3p  = (half_t*)carve((size_t)9  * F_ * F_ * 2);
  half_t* wb5p  = (half_t*)carve((size_t)25 * F_ * F_ * 2);
  half_t* wb7p  = (half_t*)carve((size_t)49 * F_ * F_ * 2);
  float*  meanb = (float*) carve(F_ * 4);
  float*  varb  = (float*) carve(F_ * 4);
  float*  ybar  = (float*) carve(B_ * F_ * 4);
  float*  yse   = (float*) carve(B_ * F_ * 4);
  (void)in_sizes; (void)n_in; (void)out_size; (void)ws_size;

  // ---- weight repack to fp16 / per-tap layout ----
  pack_f16_kernel<<<(F_ * C3F_ + 255) / 256, 256, 0, stream>>>(w_p1, wp1h, F_ * C3F_);
  pack_f16_kernel<<<(F_ * C3F_ + 255) / 256, 256, 0, stream>>>(wbp,  wbph, F_ * C3F_);
  pack_conv_kernel<<<(9  * F_ * F_ + 255) / 256, 256, 0, stream>>>(wb3, wb3p, 3);
  pack_conv_kernel<<<(25 * F_ * F_ + 255) / 256, 256, 0, stream>>>(wb5, wb5p, 5);
  pack_conv_kernel<<<(49 * F_ * F_ + 255) / 256, 256, 0, stream>>>(wb7, wb7p, 7);

  // ---- preprocess (bilateral + gamma blend) ----
  preprocess_kernel<<<P_ / 256, 256, 0, stream>>>(x, xp);

  // ---- first block: 1ch -> 192ch + relu (direct, cheap) ----
  conv_in_kernel<<<((long)P_ * C3F_ + 255) / 256, 256, 0, stream>>>(
      xp, w_i3, w_i5, w_i7, buf192);

  // ---- 1x1 conv 192->64 (WMMA GEMM), fp16 out ----
  gemm_pw_kernel<<<P_ / 16, 128, 0, stream>>>(buf192, wp1h, h2, C3F_, 1);

  // ---- second block: three implicit-GEMM WMMA convs 64->64 + relu ----
  conv_gemm_kernel<<<P_ / 16, 128, 0, stream>>>(h2, wb3p, buf192, 3, 0);
  conv_gemm_kernel<<<P_ / 16, 128, 0, stream>>>(h2, wb5p, buf192, 5, 64);
  conv_gemm_kernel<<<P_ / 16, 128, 0, stream>>>(h2, wb7p, buf192, 7, 128);

  // ---- 1x1 conv 192->64 (WMMA GEMM), fp32 out for BN stats ----
  gemm_pw_kernel<<<P_ / 16, 128, 0, stream>>>(buf192, wbph, m32, C3F_, 0);

  // ---- batchnorm (population stats over B,H,W) + relu -> fp16 ----
  bn_stats_kernel<<<F_, 256, 0, stream>>>(m32, meanb, varb);
  bn_apply_kernel<<<((long)P_ * F_ + 255) / 256, 256, 0, stream>>>(
      m32, meanb, varb, bn_g, bn_b, mbn);

  // ---- squeeze-excite ----
  se_mean_kernel<<<B_ * F_, 256, 0, stream>>>(mbn, ybar);
  se_mlp_kernel<<<1, 128, 0, stream>>>(ybar, se1, se2, yse);

  // ---- scale, final 3x3 conv 64->1, residual ----
  final_kernel<<<P_ / 256, 256, 0, stream>>>(xp, mbn, yse, wf, out);
}
